// QuestionEmbedding_17420387352929
// MI455X (gfx1250) — matrix operand order repaired and checked by reading, not compiled, not run on hardware
//
#include <hip/hip_runtime.h>

// ---------------------------------------------------------------------------
// GRU (Keras v2, reset_after=True)  B=256, T=256, D=1024, H=1024  on gfx1250
//   Phase 0: convert X -> bf16, W/U -> transposed bf16 (K-contiguous rows)
//   Phase 1: xg = X @ W + b0   (bf16 WMMA, 128x128 tiles)
//   Phase 2: 256 fused steps: rec = h@U + b1 ; gates ; h_t -> out (+bf16 copy)
// U (6 MB bf16) stays resident in the 192 MB L2 across the whole scan.
// Inner-loop staging: GLOBAL_LOAD_ASYNC_TO_LDS_B128 into double-buffered LDS
// (prefetch distance 1), so copies overlap the WMMA work of the current tile.
// ---------------------------------------------------------------------------

typedef __bf16 bf16_t;
typedef __bf16 v16bf __attribute__((ext_vector_type(16)));
typedef __bf16 v8bf  __attribute__((ext_vector_type(8)));
typedef float  v8f   __attribute__((ext_vector_type(8)));

#define GRU_H   1024
#define GRU_H3  3072
#define GRU_B   256
#define GRU_T   256
#define GRU_D   1024
#define GRU_BT  (GRU_B * GRU_T)

__device__ __forceinline__ bf16_t f32_to_bf16(float f) {
  unsigned u = __builtin_bit_cast(unsigned, f);
  u += 0x7fffu + ((u >> 16) & 1u);            // round-to-nearest-even
  unsigned short h = (unsigned short)(u >> 16);
  return __builtin_bit_cast(bf16_t, h);
}

__device__ __forceinline__ float fast_sigmoid(float x) {
  return 1.0f / (1.0f + __expf(-x));
}
__device__ __forceinline__ float fast_tanh(float x) {
  return 1.0f - 2.0f / (__expf(2.0f * x) + 1.0f);
}

// CDNA5 async copy: 16 bytes global -> LDS, tracked by ASYNCcnt.
__device__ __forceinline__ void async_b128(unsigned lds_off, const void* gsrc) {
  asm volatile("global_load_async_to_lds_b128 %0, %1, off"
               :: "v"(lds_off), "v"(gsrc) : "memory");
}
__device__ __forceinline__ void wait_async0() {
  asm volatile("s_wait_asynccnt 0x0" ::: "memory");
}
__device__ __forceinline__ unsigned lds_off32(const void* p) {
  return (unsigned)(unsigned long long)p;  // LDS aperture: addr[31:0]
}

// ---------------------------------------------------------------------------
// f32 -> bf16 straight copy (8 elements / thread)
// ---------------------------------------------------------------------------
__global__ void k_f32_to_bf16(const float* __restrict__ src,
                              bf16_t* __restrict__ dst, long long n) {
  long long i = ((long long)blockIdx.x * blockDim.x + threadIdx.x) * 8;
  if (i >= n) return;
  float4 f0 = *(const float4*)(src + i);
  float4 f1 = *(const float4*)(src + i + 4);
  v8bf o;
  o[0]=f32_to_bf16(f0.x); o[1]=f32_to_bf16(f0.y);
  o[2]=f32_to_bf16(f0.z); o[3]=f32_to_bf16(f0.w);
  o[4]=f32_to_bf16(f1.x); o[5]=f32_to_bf16(f1.y);
  o[6]=f32_to_bf16(f1.z); o[7]=f32_to_bf16(f1.w);
  *(v8bf*)(dst + i) = o;
}

// Transpose-convert: src[K][N] f32 -> dst[N][K] bf16 (K contiguous)
__global__ void k_convert_transpose(const float* __restrict__ src,
                                    bf16_t* __restrict__ dst,
                                    int K, int N) {
  long long idx = (long long)blockIdx.x * blockDim.x + threadIdx.x;
  if (idx >= (long long)K * N) return;
  int n = (int)(idx / K);
  int k = (int)(idx % K);
  dst[(long long)n * K + k] = f32_to_bf16(src[(long long)k * N + n]);
}

__global__ void k_zero_u32(unsigned* __restrict__ dst, int n) {
  int i = blockIdx.x * blockDim.x + threadIdx.x;
  if (i < n) dst[i] = 0u;
}

// ---------------------------------------------------------------------------
// Phase 1 GEMM: xg[65536][3072] = Xbf @ W + b0
// 256 threads / 8 waves, 128x128 tile. Wave (wvM,wvN) owns 32 rows x 64 cols
// => 2x4 accumulators, 8 WMMAs per 32-deep K step. Double-buffered LDS.
// ---------------------------------------------------------------------------
__global__ void __launch_bounds__(256)
k_xgemm(const bf16_t* __restrict__ Xbf,   // [BT, D]
        const bf16_t* __restrict__ Wt,    // [3H, D] (n-major, K contiguous)
        const float* __restrict__ bias0,  // b[0], [3H]
        float* __restrict__ xg)           // [BT, 3H]
{
  __shared__ __align__(64) bf16_t lds_A[2][128][32];
  __shared__ __align__(64) bf16_t lds_B[2][128][32];

  const int tid    = threadIdx.x;
  const int lane   = tid & 31;
  const int wv     = tid >> 5;      // 0..7
  const int wvM    = wv & 3;        // 32-row group
  const int wvN    = wv >> 2;       // 64-col group
  const int lane15 = lane & 15;
  const int laneHi = lane >> 4;

  const int mBase = blockIdx.x * 128;
  const int nBase = blockIdx.y * 128;

  v8f acc[2][4];
#pragma unroll
  for (int mt = 0; mt < 2; ++mt)
#pragma unroll
    for (int nt = 0; nt < 4; ++nt) acc[mt][nt] = (v8f){};

  // staging: each of 256 threads copies 32B of A and 32B of B per K step
  const int srow  = tid >> 1;       // 0..127
  const int shalf = tid & 1;
  const bf16_t* gA = Xbf + (long long)(mBase + srow) * GRU_D + shalf * 16;
  const bf16_t* gB = Wt  + (long long)(nBase + srow) * GRU_D + shalf * 16;
  const unsigned laA[2] = { lds_off32(&lds_A[0][srow][shalf * 16]),
                            lds_off32(&lds_A[1][srow][shalf * 16]) };
  const unsigned laB[2] = { lds_off32(&lds_B[0][srow][shalf * 16]),
                            lds_off32(&lds_B[1][srow][shalf * 16]) };

  // prologue: stage tile 0 into buffer 0
  async_b128(laA[0],      gA);
  async_b128(laA[0] + 16, gA + 8);
  async_b128(laB[0],      gB);
  async_b128(laB[0] + 16, gB + 8);
  wait_async0();
  __syncthreads();

  int buf = 0;
  for (int kk = 0; kk < GRU_D; kk += 32) {
    const int nb = buf ^ 1;
    // prefetch next tile into the other buffer (in flight during compute)
    if (kk + 32 < GRU_D) {
      async_b128(laA[nb],      gA + kk + 32);
      async_b128(laA[nb] + 16, gA + kk + 40);
      async_b128(laB[nb],      gB + kk + 32);
      async_b128(laB[nb] + 16, gB + kk + 40);
    }

    v16bf av[2];
#pragma unroll
    for (int mt = 0; mt < 2; ++mt) {
      const int row = wvM * 32 + mt * 16 + lane15;
      v8bf a0 = *(const v8bf*)&lds_A[buf][row][laneHi * 16];
      v8bf a1 = *(const v8bf*)&lds_A[buf][row][laneHi * 16 + 8];
      av[mt] = __builtin_shufflevector(a0, a1,
                 0,1,2,3,4,5,6,7,8,9,10,11,12,13,14,15);
    }
#pragma unroll
    for (int nt = 0; nt < 4; ++nt) {
      v16bf bv = *(const v16bf*)
          &lds_B[buf][wvN * 64 + nt * 16 + lane15][laneHi * 16];
#pragma unroll
      for (int mt = 0; mt < 2; ++mt) {
        acc[mt][nt] = __builtin_amdgcn_wmma_f32_16x16x32_bf16(
            false, av[mt], false, bv, (short)0, acc[mt][nt], false, false);
      }
    }

    wait_async0();       // next-tile copies complete
    __syncthreads();     // all threads done reading buf, copies visible
    buf = nb;
  }

#pragma unroll
  for (int nt = 0; nt < 4; ++nt) {
    const int col = nBase + wvN * 64 + nt * 16 + lane15;
    const float bias = bias0[col];
#pragma unroll
    for (int mt = 0; mt < 2; ++mt) {
#pragma unroll
      for (int r = 0; r < 8; ++r) {
        const int row = mBase + wvM * 32 + mt * 16 + r + laneHi * 8;
        xg[(long long)row * GRU_H3 + col] = acc[mt][nt][r] + bias;
      }
    }
  }
}

// ---------------------------------------------------------------------------
// Phase 2: one fused GRU step. 128 threads / 4 waves, 64(batch)x64(hidden)
// tile covering ALL THREE gates (12 WMMAs per wave per K step).
// h_{t-1} staged from bf16 ping-pong buffer (pre-zeroed for t=0).
// Double-buffered LDS, prefetch distance 1.
// ---------------------------------------------------------------------------
__global__ void __launch_bounds__(128)
k_gru_step(const float* __restrict__ xg,     // [BT, 3H] (includes b0)
           const bf16_t* __restrict__ Ut,    // [3H, H] (n-major, K contig)
           const float* __restrict__ bvec,   // [2, 3H]; b1 at +3H
           const bf16_t* __restrict__ hprevb,// [B, H] bf16 h_{t-1}
           bf16_t* __restrict__ hcurb,       // [B, H] bf16 h_t (out)
           float* __restrict__ out,          // [B, T, H]
           int t)
{
  __shared__ __align__(64) bf16_t lds_A[2][64][32];
  __shared__ __align__(64) bf16_t lds_B[2][3][64][32];

  const int tid    = threadIdx.x;
  const int lane   = tid & 31;
  const int wv     = tid >> 5;
  const int lane15 = lane & 15;
  const int laneHi = lane >> 4;

  const int bBase = blockIdx.x * 64;   // batch tile  (4)
  const int nBase = blockIdx.y * 64;   // hidden tile (16)

  v8f acc[3][4];
#pragma unroll
  for (int g = 0; g < 3; ++g)
#pragma unroll
    for (int nt = 0; nt < 4; ++nt) acc[g][nt] = (v8f){};

  const int srow  = tid >> 1;   // 0..63
  const int shalf = tid & 1;
  const bf16_t* gA  = hprevb + (long long)(bBase + srow) * GRU_H + shalf * 16;
  const bf16_t* gB0 = Ut + (long long)(0 * GRU_H + nBase + srow) * GRU_H + shalf * 16;
  const bf16_t* gB1 = Ut + (long long)(1 * GRU_H + nBase + srow) * GRU_H + shalf * 16;
  const bf16_t* gB2 = Ut + (long long)(2 * GRU_H + nBase + srow) * GRU_H + shalf * 16;
  const unsigned la[2]  = { lds_off32(&lds_A[0][srow][shalf * 16]),
                            lds_off32(&lds_A[1][srow][shalf * 16]) };
  const unsigned lb0[2] = { lds_off32(&lds_B[0][0][srow][shalf * 16]),
                            lds_off32(&lds_B[1][0][srow][shalf * 16]) };
  const unsigned lb1[2] = { lds_off32(&lds_B[0][1][srow][shalf * 16]),
                            lds_off32(&lds_B[1][1][srow][shalf * 16]) };
  const unsigned lb2[2] = { lds_off32(&lds_B[0][2][srow][shalf * 16]),
                            lds_off32(&lds_B[1][2][srow][shalf * 16]) };

  // prologue: stage tile 0 into buffer 0
  async_b128(la[0],       gA);
  async_b128(la[0]  + 16, gA + 8);
  async_b128(lb0[0],      gB0);
  async_b128(lb0[0] + 16, gB0 + 8);
  async_b128(lb1[0],      gB1);
  async_b128(lb1[0] + 16, gB1 + 8);
  async_b128(lb2[0],      gB2);
  async_b128(lb2[0] + 16, gB2 + 8);
  wait_async0();
  __syncthreads();

  int buf = 0;
  for (int kk = 0; kk < GRU_H; kk += 32) {
    const int nb = buf ^ 1;
    if (kk + 32 < GRU_H) {
      async_b128(la[nb],       gA  + kk + 32);
      async_b128(la[nb]  + 16, gA  + kk + 40);
      async_b128(lb0[nb],      gB0 + kk + 32);
      async_b128(lb0[nb] + 16, gB0 + kk + 40);
      async_b128(lb1[nb],      gB1 + kk + 32);
      async_b128(lb1[nb] + 16, gB1 + kk + 40);
      async_b128(lb2[nb],      gB2 + kk + 32);
      async_b128(lb2[nb] + 16, gB2 + kk + 40);
    }

    v8bf a0 = *(const v8bf*)&lds_A[buf][wv * 16 + lane15][laneHi * 16];
    v8bf a1 = *(const v8bf*)&lds_A[buf][wv * 16 + lane15][laneHi * 16 + 8];
    v16bf av = __builtin_shufflevector(a0, a1,
                 0,1,2,3,4,5,6,7,8,9,10,11,12,13,14,15);
#pragma unroll
    for (int g = 0; g < 3; ++g) {
#pragma unroll
      for (int nt = 0; nt < 4; ++nt) {
        v16bf bv = *(const v16bf*)
            &lds_B[buf][g][nt * 16 + lane15][laneHi * 16];
        acc[g][nt] = __builtin_amdgcn_wmma_f32_16x16x32_bf16(
            false, av, false, bv, (short)0, acc[g][nt], false, false);
      }
    }

    wait_async0();
    __syncthreads();
    buf = nb;
  }

  // ---- gates + output (f32 to out, bf16 to ping-pong h buffer)
#pragma unroll
  for (int nt = 0; nt < 4; ++nt) {
    const int j = nBase + nt * 16 + lane15;
    const float b1z = bvec[GRU_H3 + j];
    const float b1r = bvec[GRU_H3 + GRU_H + j];
    const float b1h = bvec[GRU_H3 + 2 * GRU_H + j];
#pragma unroll
    for (int r = 0; r < 8; ++r) {
      const int b_ = bBase + wv * 16 + r + laneHi * 8;
      const long long rowt = (long long)b_ * GRU_T + t;
      const float* xrow = xg + rowt * GRU_H3;
      const float xz = xrow[j];
      const float xr = xrow[GRU_H + j];
      const float xh = xrow[2 * GRU_H + j];
      const float hprev = (t > 0)
          ? out[((long long)b_ * GRU_T + (t - 1)) * GRU_H + j] : 0.0f;
      const float z  = fast_sigmoid(xz + acc[0][nt][r] + b1z);
      const float rr = fast_sigmoid(xr + acc[1][nt][r] + b1r);
      const float hh = fast_tanh(xh + rr * (acc[2][nt][r] + b1h));
      const float hnew = z * hprev + (1.0f - z) * hh;
      out[rowt * GRU_H + j] = hnew;
      hcurb[(long long)b_ * GRU_H + j] = f32_to_bf16(hnew);
    }
  }
}

// ---------------------------------------------------------------------------
extern "C" void kernel_launch(void* const* d_in, const int* in_sizes, int n_in,
                              void* d_out, int out_size, void* d_ws, size_t ws_size,
                              hipStream_t stream) {
  (void)in_sizes; (void)n_in; (void)out_size; (void)ws_size;
  const float* question = (const float*)d_in[0];  // [B, T, D]
  const float* W        = (const float*)d_in[1];  // [D, 3H]
  const float* U        = (const float*)d_in[2];  // [H, 3H]
  const float* b        = (const float*)d_in[3];  // [2, 3H]
  float* out = (float*)d_out;                     // [B, T, H]

  // ws: xg f32 | Xbf bf16 | Wt bf16 | Ut bf16 | hbf bf16 [2][B][H]
  float*  xg = (float*)d_ws;
  size_t  xgElems = (size_t)GRU_BT * GRU_H3;
  bf16_t* Xbf = (bf16_t*)(xg + xgElems);
  bf16_t* Wt  = Xbf + (size_t)GRU_BT * GRU_D;
  bf16_t* Ut  = Wt + (size_t)GRU_H3 * GRU_D;
  bf16_t* hbf = Ut + (size_t)GRU_H3 * GRU_H;

  {  // X -> bf16
    long long n = (long long)GRU_BT * GRU_D;
    int blocks = (int)((n / 8 + 255) / 256);
    k_f32_to_bf16<<<blocks, 256, 0, stream>>>(question, Xbf, n);
  }
  {  // W, U -> transposed bf16
    long long total = (long long)GRU_H3 * GRU_D;
    int blocks = (int)((total + 255) / 256);
    k_convert_transpose<<<blocks, 256, 0, stream>>>(W, Wt, GRU_D, GRU_H3);
    k_convert_transpose<<<blocks, 256, 0, stream>>>(U, Ut, GRU_H, GRU_H3);
  }
  {  // zero the t=-1 ping-pong h buffer: index (0-1)&1 == 1
    int nWords = GRU_B * GRU_H / 2;  // bf16 pairs
    k_zero_u32<<<(nWords + 255) / 256, 256, 0, stream>>>(
        (unsigned*)(hbf + (size_t)GRU_B * GRU_H), nWords);
  }

  k_xgemm<<<dim3(GRU_BT / 128, GRU_H3 / 128), 256, 0, stream>>>(
      Xbf, Wt, b, xg);

  for (int t = 0; t < GRU_T; ++t) {
    bf16_t* hprev = hbf + (size_t)((t - 1) & 1) * GRU_B * GRU_H;
    bf16_t* hcur  = hbf + (size_t)(t & 1) * GRU_B * GRU_H;
    k_gru_step<<<dim3(GRU_B / 64, GRU_H / 64), 128, 0, stream>>>(
        xg, Ut, b, hprev, hcur, out, t);
  }
}
